// MSDeformAttn_5806795784257
// MI455X (gfx1250) — matrix operand order repaired and checked by reading, not compiled, not run on hardware
//
#include <hip/hip_runtime.h>
#include <hip/hip_bf16.h>
#include <math.h>

typedef __attribute__((ext_vector_type(16))) _Float16 v16h;
typedef __attribute__((ext_vector_type(8)))  _Float16 v8h;
typedef __attribute__((ext_vector_type(8)))  float    v8f;
typedef __attribute__((ext_vector_type(4)))  unsigned int v4u;
typedef __attribute__((ext_vector_type(8)))  int      v8i;
typedef __attribute__((ext_vector_type(4)))  int      v4i;

// ---------------------------------------------------------------------------
// f32 -> f16 conversion (one-time pass so GEMMs can run on the WMMA path)
// ---------------------------------------------------------------------------
__global__ __launch_bounds__(256) void cvt_f32_f16(const float* __restrict__ in,
                                                   _Float16* __restrict__ out,
                                                   int n) {
    int i = blockIdx.x * blockDim.x + threadIdx.x;
    if (i < n) out[i] = (_Float16)in[i];
}

// ---------------------------------------------------------------------------
// WMMA GEMM:  Out[r, c] = sum_k A[r, k] * W[c, k] + bias[c]
//   A : rows x K (f16, row-major)     W : cols x K (f16, row-major)
//   Out : rows x cols (f32)
//
// The whole W panel (cols x K f16, <=128 KB) is staged ONCE per block into
// dynamic LDS by the Tensor Data Mover (tensor_load_to_lds, TENSORcnt), then
// all 8 waves read their B fragments from LDS (ds_load_b128 pairs) while the
// A fragments stream from global. Each wave owns a 16-row x 64-col slab
// (4 f32 accumulators, A fragment reused 4x per k-step).
// K multiple of 32; cols multiple of 64 and <= 256.
// ---------------------------------------------------------------------------
__global__ __launch_bounds__(256) void gemm_wmma_f16(
    const _Float16* __restrict__ A,
    const _Float16* __restrict__ W,
    const float*    __restrict__ bias,
    float*          __restrict__ Out,
    int rows, int cols, int K)
{
    extern __shared__ _Float16 ldsW[];            // cols * K halves at LDS offset 0

    const int wave = threadIdx.x >> 5;
    const int lane = threadIdx.x & 31;
    const int hl   = lane & 15;                   // half-lane (0..15)
    const int hi   = lane >> 4;                   // lane group (0 or 1)

    const int slabs = cols >> 6;                  // 64-col slabs (4 or 2)
    const int rowTilesPerBlock = 8 / slabs;
    const int tilesM = (rows + 15) >> 4;

    // ---- Stage W panel into LDS -------------------------------------------
#if __has_builtin(__builtin_amdgcn_tensor_load_to_lds)
    {
        // Tensor DMA descriptor (ISA 08_async_tensor §8): 2-D tensor of
        // 2-byte elements, dim0 = K (contiguous), dim1 = cols, tile == tensor.
        const unsigned long long waddr = (unsigned long long)W;
        v4u g0 = { 1u,                                   // count=1, user mode
                   0u,                                   // lds_addr = dynamic-LDS base
                   (unsigned int)(waddr & 0xFFFFFFFFu),  // global_addr[31:0]
                   (unsigned int)((waddr >> 32) & 0x01FFFFFFu) | 0x80000000u }; // addr[56:32] | type=2
        v8i g1 = { (int)(1u << 16),                                   // data_size=1 (2B), mask=0
                   (int)((unsigned)(K & 0xFFFF) << 16),               // tensor_dim0[15:0]
                   (int)(((unsigned)(K >> 16) & 0xFFFFu) |
                         ((unsigned)(cols & 0xFFFF) << 16)),          // dim0[31:16] | dim1[15:0]
                   (int)(((unsigned)(cols >> 16) & 0xFFFFu) |
                         ((unsigned)(K & 0xFFFF) << 16)),             // dim1[31:16] | tile_dim0
                   (int)(cols & 0xFFFF),                              // tile_dim1 (tile_dim2=0)
                   (int)K,                                            // dim0_stride[31:0]
                   0, 0 };                                            // stride hi / dim1_stride
        v4i g2 = { 0, 0, 0, 0 };                  // 2-D: groups 2/3 unused
        v4i g3 = { 0, 0, 0, 0 };
        if (wave == 0) {                          // one TDM issue per block (EXEC ignored)
#if __clang_major__ >= 23
            v8i g4 = { 0, 0, 0, 0, 0, 0, 0, 0 };
            __builtin_amdgcn_tensor_load_to_lds(g0, g1, g2, g3, g4, 0);
#else
            __builtin_amdgcn_tensor_load_to_lds(g0, g1, g2, g3, 0);
#endif
        }
        __builtin_amdgcn_s_wait_tensorcnt(0);     // TENSORcnt==0 (nonzero only in wave 0)
    }
#else
    {   // fallback: cooperative global->LDS copy
        const int totalVec = (cols * K) >> 3;     // 16B chunks
        for (int i = threadIdx.x; i < totalVec; i += blockDim.x)
            ((v8h*)ldsW)[i] = ((const v8h*)W)[i];
    }
#endif
    __syncthreads();

    // ---- Compute -----------------------------------------------------------
    const int tm = blockIdx.x * rowTilesPerBlock + (wave / slabs);
    const int tn = wave % slabs;
    if (tm >= tilesM) return;                     // after the barrier: safe

    const int mBase = tm << 4;
    const int nBase = tn << 6;

    int m = mBase + hl;                           // A row (ISA 7.12.2 A layout)
    if (m >= rows) m = rows - 1;                  // clamp; guarded on store
    const _Float16* Arow = A + (size_t)m * K;

    v8f acc0 = {}, acc1 = {}, acc2 = {}, acc3 = {};

    for (int k0 = 0; k0 < K; k0 += 32) {
        // A fragment: lane group hi holds K = hi*16..+15
        v8h a0 = *(const v8h*)(Arow + k0 + hi * 16);
        v8h a1 = *(const v8h*)(Arow + k0 + hi * 16 + 8);
        v16h a;
#pragma unroll
        for (int j = 0; j < 8; ++j) { a[j] = a0[j]; a[8 + j] = a1[j]; }

        // B fragment t from LDS: column n = nBase+t*16+hl holds K = hi*16..+15
        // (B[k][n] = W[n][k] -> contiguous 32B run of LDS row n)
        v16h b0 = *(const v16h*)(ldsW + (size_t)(nBase +  0 + hl) * K + k0 + hi * 16);
        v16h b1 = *(const v16h*)(ldsW + (size_t)(nBase + 16 + hl) * K + k0 + hi * 16);
        v16h b2 = *(const v16h*)(ldsW + (size_t)(nBase + 32 + hl) * K + k0 + hi * 16);
        v16h b3 = *(const v16h*)(ldsW + (size_t)(nBase + 48 + hl) * K + k0 + hi * 16);

        acc0 = __builtin_amdgcn_wmma_f32_16x16x32_f16(false, a, false, b0, (short)0, acc0, false, false);
        acc1 = __builtin_amdgcn_wmma_f32_16x16x32_f16(false, a, false, b1, (short)0, acc1, false, false);
        acc2 = __builtin_amdgcn_wmma_f32_16x16x32_f16(false, a, false, b2, (short)0, acc2, false, false);
        acc3 = __builtin_amdgcn_wmma_f32_16x16x32_f16(false, a, false, b3, (short)0, acc3, false, false);
    }

    // C/D layout: VGPR r -> row (hi*8 + r), lane -> col hl.
    v8f accs[4] = {acc0, acc1, acc2, acc3};
#pragma unroll
    for (int t = 0; t < 4; ++t) {
        const int n  = nBase + t * 16 + hl;
        const float bn = bias[n];
#pragma unroll
        for (int r = 0; r < 8; ++r) {
            const int mm = mBase + hi * 8 + r;
            if (mm < rows) Out[(size_t)mm * cols + n] = accs[t][r] + bn;
        }
    }
}

// ---------------------------------------------------------------------------
// Softmax over 16 logits per (n, q, head), in place.
// ---------------------------------------------------------------------------
__global__ __launch_bounds__(256) void softmax16(float* __restrict__ attn, int nrows) {
    int r = blockIdx.x * blockDim.x + threadIdx.x;
    if (r >= nrows) return;
    float* p = attn + (size_t)r * 16;
    float mx = p[0];
#pragma unroll
    for (int i = 1; i < 16; ++i) mx = fmaxf(mx, p[i]);
    float s = 0.f;
    float e[16];
#pragma unroll
    for (int i = 0; i < 16; ++i) { e[i] = __expf(p[i] - mx); s += e[i]; }
    const float inv = 1.f / s;
#pragma unroll
    for (int i = 0; i < 16; ++i) p[i] = e[i] * inv;
}

// ---------------------------------------------------------------------------
// Deformable sampling: one wave per (n*Lq + q, head m); lane = channel d.
// value (27 MB) is L2-resident (192 MB), so 4-corner gathers are L2 hits;
// each corner fetch is a coalesced 32-lane x 4B burst.
// Writes sampled activations as f16 (A operand for the final WMMA GEMM).
// ---------------------------------------------------------------------------
__global__ __launch_bounds__(256) void msda_sample(
    const float*     __restrict__ ref,     // NQ x L x 2
    const float*     __restrict__ off,     // NQ x 256  (m,l,p,xy)
    const float*     __restrict__ attn,    // NQ x 8 x 16 (softmaxed)
    const float*     __restrict__ value,   // (N*LenIn) x 256, c = m*32+d
    const long long* __restrict__ shapes,  // L x 2 (H, W)
    const long long* __restrict__ starts,  // L
    _Float16*        __restrict__ sampled, // NQ x 256
    int NQ, int Lq, int LenIn)
{
    const int wavesPerBlock = blockDim.x >> 5;
    const int gwave = blockIdx.x * wavesPerBlock + (threadIdx.x >> 5);
    const int lane  = threadIdx.x & 31;          // channel d within head
    const int nq = gwave >> 3;                   // query index (n*Lq + q)
    const int m  = gwave & 7;                    // head
    if (nq >= NQ) return;
    const int n = nq / Lq;

    const float* offp = off  + (size_t)nq * 256 + m * 32;       // l*8+p*2+xy
    const float* attp = attn + ((size_t)nq * 8 + m) * 16;       // l*4+p
    float acc = 0.f;

    for (int l = 0; l < 4; ++l) {
        const int H  = (int)shapes[l * 2 + 0];
        const int W  = (int)shapes[l * 2 + 1];
        const int s0 = (int)starts[l];
        const float rx = ref[((size_t)nq * 4 + l) * 2 + 0];
        const float ry = ref[((size_t)nq * 4 + l) * 2 + 1];
        const float* vb = value + ((size_t)n * LenIn + s0) * 256 + m * 32 + lane;
        const float invW = 1.f / (float)W, invH = 1.f / (float)H;

        for (int p = 0; p < 4; ++p) {
            const float ox = offp[l * 8 + p * 2 + 0];
            const float oy = offp[l * 8 + p * 2 + 1];
            // grids = 2*loc-1 ; gx = ((grid+1)*W-1)/2 = loc*W - 0.5
            const float gx = (rx + ox * invW) * (float)W - 0.5f;
            const float gy = (ry + oy * invH) * (float)H - 0.5f;
            const float x0f = floorf(gx), y0f = floorf(gy);
            const float fx = gx - x0f, fy = gy - y0f;
            const int x0 = (int)x0f, y0 = (int)y0f;
            const float aw = attp[l * 4 + p];

            const float w00 = (1.f - fy) * (1.f - fx) * aw;
            const float w01 = (1.f - fy) * fx * aw;
            const float w10 = fy * (1.f - fx) * aw;
            const float w11 = fy * fx * aw;

            const bool vy0 = (y0 >= 0) & (y0 < H);
            const bool vy1 = (y0 + 1 >= 0) & (y0 + 1 < H);
            const bool vx0 = (x0 >= 0) & (x0 < W);
            const bool vx1 = (x0 + 1 >= 0) & (x0 + 1 < W);

            if (vy0 & vx0) acc += w00 * vb[(size_t)(y0 * W + x0) * 256];
            if (vy0 & vx1) acc += w01 * vb[(size_t)(y0 * W + x0 + 1) * 256];
            if (vy1 & vx0) acc += w10 * vb[(size_t)((y0 + 1) * W + x0) * 256];
            if (vy1 & vx1) acc += w11 * vb[(size_t)((y0 + 1) * W + x0 + 1) * 256];
        }
    }
    sampled[(size_t)nq * 256 + m * 32 + lane] = (_Float16)acc;
}

// ---------------------------------------------------------------------------
// Host-side orchestration
// ---------------------------------------------------------------------------
static inline int cdiv(int a, int b) { return (a + b - 1) / b; }
static inline size_t align256(size_t x) { return (x + 255) & ~(size_t)255; }

static void launch_gemm(const _Float16* A, const _Float16* W, const float* bias,
                        float* Out, int rows, int cols, int K, hipStream_t stream) {
    const int slabs = cols / 64;
    const int rtpb  = 8 / slabs;                    // row tiles per block
    const int tilesM = (rows + 15) / 16;
    const int grid = cdiv(tilesM, rtpb);
    const size_t ldsBytes = (size_t)cols * K * sizeof(_Float16);  // <= 128 KB (WGP has 320 KB)
    gemm_wmma_f16<<<grid, 256, ldsBytes, stream>>>(A, W, bias, Out, rows, cols, K);
}

extern "C" void kernel_launch(void* const* d_in, const int* in_sizes, int n_in,
                              void* d_out, int out_size, void* d_ws, size_t ws_size,
                              hipStream_t stream) {
    const float*     query  = (const float*)d_in[0];
    const float*     ref    = (const float*)d_in[1];
    const float*     inpf   = (const float*)d_in[2];
    const long long* shapes = (const long long*)d_in[3];
    const long long* starts = (const long long*)d_in[4];
    const float*     Wv     = (const float*)d_in[5];
    const float*     bv     = (const float*)d_in[6];
    const float*     Woff   = (const float*)d_in[7];
    const float*     boff   = (const float*)d_in[8];
    const float*     Wa     = (const float*)d_in[9];
    const float*     ba     = (const float*)d_in[10];
    const float*     Wo     = (const float*)d_in[11];
    const float*     bo     = (const float*)d_in[12];

    const int C  = 256;
    const int N  = 2;
    const int NQ = in_sizes[0] / C;      // N * Lq
    const int NV = in_sizes[2] / C;      // N * Len_in
    const int Lq = NQ / N;
    const int LenIn = NV / N;

    // Workspace carve-up (all offsets 256B aligned)
    char* ws = (char*)d_ws;
    size_t o = 0;
    _Float16* qh    = (_Float16*)(ws + o); o = align256(o + (size_t)NQ * C * 2);
    _Float16* ih    = (_Float16*)(ws + o); o = align256(o + (size_t)NV * C * 2);
    _Float16* Wvh   = (_Float16*)(ws + o); o = align256(o + (size_t)C * C * 2);
    _Float16* Woffh = (_Float16*)(ws + o); o = align256(o + (size_t)256 * C * 2);
    _Float16* Wah   = (_Float16*)(ws + o); o = align256(o + (size_t)128 * C * 2);
    _Float16* Woh   = (_Float16*)(ws + o); o = align256(o + (size_t)C * C * 2);
    float*    valf  = (float*)   (ws + o); o = align256(o + (size_t)NV * C * 4);
    float*    offf  = (float*)   (ws + o); o = align256(o + (size_t)NQ * 256 * 4);
    float*    attnf = (float*)   (ws + o); o = align256(o + (size_t)NQ * 128 * 4);
    _Float16* samph = (_Float16*)(ws + o); o = align256(o + (size_t)NQ * C * 2);
    (void)ws_size;

    // 1) precision conversion passes
    cvt_f32_f16<<<cdiv(NQ * C, 256), 256, 0, stream>>>(query, qh, NQ * C);
    cvt_f32_f16<<<cdiv(NV * C, 256), 256, 0, stream>>>(inpf, ih, NV * C);
    cvt_f32_f16<<<cdiv(C * C, 256), 256, 0, stream>>>(Wv, Wvh, C * C);
    cvt_f32_f16<<<cdiv(256 * C, 256), 256, 0, stream>>>(Woff, Woffh, 256 * C);
    cvt_f32_f16<<<cdiv(128 * C, 256), 256, 0, stream>>>(Wa, Wah, 128 * C);
    cvt_f32_f16<<<cdiv(C * C, 256), 256, 0, stream>>>(Wo, Woh, C * C);

    // 2) value = input_flatten @ Wv.T + bv     (NV x 256, K=256)
    launch_gemm(ih, Wvh, bv, valf, NV, 256, C, stream);
    // 3) off = query @ Woff.T + boff           (NQ x 256)
    launch_gemm(qh, Woffh, boff, offf, NQ, 256, C, stream);
    // 4) attn logits = query @ Wa.T + ba       (NQ x 128)
    launch_gemm(qh, Wah, ba, attnf, NQ, 128, C, stream);

    // 5) softmax over 16 per (query, head)
    softmax16<<<cdiv(NQ * 8, 256), 256, 0, stream>>>(attnf, NQ * 8);

    // 6) deformable bilinear sampling -> f16 activations
    {
        int waves = NQ * 8;                 // one wave per (query, head)
        msda_sample<<<cdiv(waves, 8), 256, 0, stream>>>(
            ref, offf, attnf, valf, shapes, starts, samph, NQ, Lq, LenIn);
    }

    // 7) out = sampled @ Wo.T + bo -> d_out (f32)
    launch_gemm(samph, Woh, bo, (float*)d_out, NQ, 256, C, stream);
}
